// GRU_D_42004780155213
// MI455X (gfx1250) — matrix operand order repaired and checked by reading, not compiled, not run on hardware
//
#include <hip/hip_runtime.h>
#include <hip/hip_bf16.h>

typedef __attribute__((ext_vector_type(16))) _Float16 v16h;
typedef __attribute__((ext_vector_type(8)))  float    v8f;
typedef __attribute__((ext_vector_type(4)))  float    v4f;

#define NF   64
#define LH   96
#define LP   24
#define LALL 120
#define HH   128
#define BSZ  1024
#define NTHREADS 384   // 12 waves: two 16-col j-tiles per wave, weights in VGPRs

// ---------------------------------------------------------------------------
// Prep kernel 1: LOCF scan, mu (masked mean incl. zero-pad region), dt.
// One block per batch row, 128 threads.
// ---------------------------------------------------------------------------
__global__ void grud_prep_kernel(const float* __restrict__ tp_pred,
                                 const float* __restrict__ X,
                                 const float* __restrict__ tp_true,
                                 const float* __restrict__ mask,
                                 float* __restrict__ locf,
                                 float* __restrict__ dtb,
                                 float* __restrict__ mu) {
    const int b = blockIdx.x;
    const int tid = threadIdx.x;
    if (tid < NF) {
        const int n = tid;
        float prev = X[((size_t)b * LH) * NF + n];
        float sum = 0.f, cnt = 0.f;
        for (int t = 0; t < LH; ++t) {
            float x  = X[((size_t)b * LH + t) * NF + n];
            float mv = mask[((size_t)b * LH + t) * NF + n];
            float valid = 1.f - mv;
            sum += x * valid;
            cnt += valid;
            float cur = (mv != 0.f) ? prev : x;     // reference scan semantics
            locf[((size_t)b * LALL + t) * NF + n] = cur;
            prev = cur;
        }
        // pad region: x=0, m=0 -> cur = x = 0 ; valid=1 contributes to cnt
        for (int t = LH; t < LALL; ++t)
            locf[((size_t)b * LALL + t) * NF + n] = 0.f;
        cnt += (float)LP;
        mu[(size_t)b * NF + n] = sum / fmaxf(cnt, 1.f);
    }
    if (tid < LALL) {
        const int t = tid;
        float d = 0.f;
        if (t > 0) {
            float cur = (t     < LH) ? tp_true[(size_t)b * LH + t]
                                     : tp_pred[(size_t)b * LP + (t - LH)];
            float prv = (t - 1 < LH) ? tp_true[(size_t)b * LH + (t - 1)]
                                     : tp_pred[(size_t)b * LP + (t - 1 - LH)];
            d = cur - prv;
        }
        dtb[(size_t)b * LALL + t] = d;
    }
}

// ---------------------------------------------------------------------------
// Prep kernel 2: pack W_ih^T / W_hh^T / Wp^T into per-lane WMMA B-fragment
// order (f16, 16 contiguous halves = 32B per lane), plus rowsum(Wh_dec).
// B layout (16-bit, 32x16 KxN): lane = hi*16 + n_lo ; k = kt*32 + (e&7) + hi*8 + (e>>3)*16.
// ---------------------------------------------------------------------------
#define NIH (8 * 24 * 512)    // W_ih: 8 k-tiles x 24 j-tiles x 32 lanes x 16 halves
#define NHH (4 * 24 * 512)    // W_hh: 4 k-tiles x 24 j-tiles
#define NP  (4 * 4 * 512)     // Wp  : 4 k-tiles x 4  j-tiles

__global__ void grud_pack_kernel(const float* __restrict__ W_ih,
                                 const float* __restrict__ W_hh,
                                 const float* __restrict__ Wp,
                                 const float* __restrict__ Wh_dec,
                                 _Float16* __restrict__ fragIH,
                                 _Float16* __restrict__ fragHH,
                                 _Float16* __restrict__ fragP,
                                 float* __restrict__ rowsumWh) {
    const int idx = blockIdx.x * blockDim.x + threadIdx.x;
    if (idx < NIH) {
        int kt = idx / (24 * 512), rem = idx % (24 * 512);
        int jt = rem / 512, l = (rem % 512) / 16, e = rem % 16;
        int lo = l & 15, hi = l >> 4;
        int j = jt * 16 + lo, k = kt * 32 + (e & 7) + hi * 8 + (e >> 3) * 16;
        fragIH[idx] = (_Float16)W_ih[j * 256 + k];
    } else if (idx < NIH + NHH) {
        int f = idx - NIH;
        int kt = f / (24 * 512), rem = f % (24 * 512);
        int jt = rem / 512, l = (rem % 512) / 16, e = rem % 16;
        int lo = l & 15, hi = l >> 4;
        int j = jt * 16 + lo, k = kt * 32 + (e & 7) + hi * 8 + (e >> 3) * 16;
        fragHH[f] = (_Float16)W_hh[j * HH + k];
    } else if (idx < NIH + NHH + NP) {
        int f = idx - NIH - NHH;
        int kt = f / (4 * 512), rem = f % (4 * 512);
        int jt = rem / 512, l = (rem % 512) / 16, e = rem % 16;
        int lo = l & 15, hi = l >> 4;
        int j = jt * 16 + lo, k = kt * 32 + (e & 7) + hi * 8 + (e >> 3) * 16;
        fragP[f] = (_Float16)Wp[j * HH + k];
    } else if (idx < NIH + NHH + NP + HH) {
        int hh = idx - (NIH + NHH + NP);
        float s = 0.f;
        for (int n = 0; n < NF; ++n) s += Wh_dec[hh * NF + n];
        rowsumWh[hh] = s;
    }
}

// ---------------------------------------------------------------------------
// A-fragment pack helper. 16-bit A 16x32 layout (ISA 7.12.2):
// lane = hi*16 + m ; k' = (e&7) + (e>>3)*16 + hi*8  -> inverse below.
// ---------------------------------------------------------------------------
__device__ __forceinline__ void packA(_Float16* sA, int m, int k, float v) {
    int kt = k >> 5, kp = k & 31;
    int hi = (kp >> 3) & 1;
    int e  = (kp & 7) | (((kp >> 4) & 1) << 3);
    sA[kt * 512 + (hi * 16 + m) * 16 + e] = (_Float16)v;
}

__device__ __forceinline__ float sigmoidf_fast(float x) {
    return 1.f / (1.f + __expf(-x));
}
__device__ __forceinline__ float tanhf_fast(float x) {
    return 1.f - 2.f / (__expf(2.f * x) + 1.f);
}

// ---------------------------------------------------------------------------
// Main GRU-D kernel: 64 blocks x 384 threads (12 waves, two j-tiles each).
// All W_ih / W_hh B-fragments are pre-loaded into VGPRs before the time loop
// (24 x v16h = 192 VGPRs/wave; 3 waves/SIMD gives a ~336 VGPR budget), so the
// steady-state GEMM touches no global/scratch memory at all. Each block owns
// a 16-row batch tile and runs all 120 steps sequentially using
// v_wmma_f32_16x16x32_f16 with f32 accumulation.
// ---------------------------------------------------------------------------
__global__ void __launch_bounds__(NTHREADS, 1)
grud_main_kernel(const float* __restrict__ X,
                 const float* __restrict__ mask,
                 const float* __restrict__ bh_dec,
                 const float* __restrict__ Wx_dec,
                 const float* __restrict__ bx_dec,
                 const float* __restrict__ b_ih,
                 const float* __restrict__ b_hh,
                 const float* __restrict__ bp,
                 const float* __restrict__ locf,
                 const float* __restrict__ dtb,
                 const float* __restrict__ mu,
                 const float* __restrict__ rowsumWh,
                 const _Float16* __restrict__ fragIH,
                 const _Float16* __restrict__ fragHH,
                 const _Float16* __restrict__ fragP,
                 float* __restrict__ out) {
    __shared__ __align__(32) _Float16 sA[8 * 512];   // inp = [x_in | h | m], A-fragment order
    __shared__ __align__(32) float sS[256 * 16];     // (gi+gr) r/z blocks, column-major [j][m]
    __shared__ __align__(32) float sGiN[HH * 16];    // gi n-block, column-major [col][m]
    __shared__ __align__(32) float sGrN[HH * 16];    // gr n-block, column-major [col][m]
    __shared__ float sH[16 * HH];                    // fp32 hidden state, row-major [m][h]
    __shared__ float sMu[16 * NF];
    __shared__ float sDt[16];
    __shared__ float sBih[384], sBhh[384];
    __shared__ float sBhDec[HH], sRow[HH];
    __shared__ float sBxDec[NF], sDiag[NF], sBp[NF];

    const int tid  = threadIdx.x;
    const int wv   = tid >> 5;       // 0..11
    const int lane = tid & 31;
    const int b0   = blockIdx.x * 16;

    // ---- one-time loads into LDS ----
    for (int i = tid; i < 384; i += NTHREADS) { sBih[i] = b_ih[i]; sBhh[i] = b_hh[i]; }
    if (tid < HH) { sBhDec[tid] = bh_dec[tid]; sRow[tid] = rowsumWh[tid]; }
    if (tid < NF) {
        sBxDec[tid] = bx_dec[tid];
        sDiag[tid]  = Wx_dec[tid * NF + tid];
        sBp[tid]    = bp[tid];
    }
    for (int i = tid; i < 16 * NF; i += NTHREADS) {
        int m = i >> 6, n = i & 63;
        sMu[i] = mu[(size_t)(b0 + m) * NF + n];
    }
    for (int i = tid; i < 16 * HH; i += NTHREADS) sH[i] = 0.f;

    // ---- pre-load this wave's B fragments into registers (loop-invariant) ----
    const int jt0 = wv * 2;          // tiles jt0 and jt0+1
    const int lo = lane & 15, hi = lane >> 4;
    v16h Bih[2][8];
    v16h Bhh[2][4];
    #pragma unroll
    for (int j = 0; j < 2; ++j) {
        const int jtx = jt0 + j;
        #pragma unroll
        for (int kt = 0; kt < 8; ++kt)
            Bih[j][kt] = *(const v16h*)(fragIH + ((size_t)kt * 24 + jtx) * 512 + lane * 16);
        #pragma unroll
        for (int kt = 0; kt < 4; ++kt)
            Bhh[j][kt] = *(const v16h*)(fragHH + ((size_t)kt * 24 + jtx) * 512 + lane * 16);
    }
    // warm the projection weights into L2 for the t>=96 branch
    __builtin_prefetch(fragP + (size_t)lane * 16, 0, 2);
    __syncthreads();

    for (int t = 0; t < LALL; ++t) {
        if (tid < 16) sDt[tid] = dtb[(size_t)(b0 + tid) * LALL + t];
        __syncthreads();   // dt ready; previous step's readers of sA done

        // ---- decay hidden state, pack h into A fragments (K = 64..191) ----
        for (int i = tid; i < 16 * HH; i += NTHREADS) {
            int m = i >> 7, hh = i & 127;
            float g  = __expf(-fmaxf(sDt[m] * sRow[hh] + sBhDec[hh], 0.f));
            float hv = sH[i] * g;
            sH[i] = hv;                 // rep = decayed h
            packA(sA, m, 64 + hh, hv);
        }
        // ---- impute x_in, pack x_in (K=0..63) and mask (K=192..255) ----
        for (int i = tid; i < 16 * NF; i += NTHREADS) {
            int m = i >> 6, n = i & 63;
            int b = b0 + m;
            float xt = 0.f, mt = 0.f;
            if (t < LH) {
                xt = X[((size_t)b * LH + t) * NF + n];
                mt = mask[((size_t)b * LH + t) * NF + n];
            }
            float gx  = __expf(-fmaxf(sDt[m] * sDiag[n] + sBxDec[n], 0.f));
            float xl  = locf[((size_t)b * LALL + t) * NF + n];
            float xh  = gx * xl + (1.f - gx) * sMu[i];
            float xin = mt * xt + (1.f - mt) * xh;
            packA(sA, m, n, xin);
            packA(sA, m, 192 + n, mt);
        }
        __syncthreads();   // A fragments + sH ready

        // ---- GEMMs: gi = inp @ W_ih^T, gr = h @ W_hh^T ----
        // Two j-tiles per wave; each A fragment is loaded once and feeds both.
        {
            v8f gi0 = {}, gi1 = {}, gr0 = {}, gr1 = {};
            #pragma unroll
            for (int kt = 0; kt < 8; ++kt) {
                v16h a = *(const v16h*)(sA + kt * 512 + lane * 16);
                gi0 = __builtin_amdgcn_wmma_f32_16x16x32_f16(false, a, false, Bih[0][kt],
                                                             (short)0, gi0, false, false);
                gi1 = __builtin_amdgcn_wmma_f32_16x16x32_f16(false, a, false, Bih[1][kt],
                                                             (short)0, gi1, false, false);
            }
            #pragma unroll
            for (int kt = 0; kt < 4; ++kt) {
                v16h a = *(const v16h*)(sA + (kt + 2) * 512 + lane * 16);  // h region
                gr0 = __builtin_amdgcn_wmma_f32_16x16x32_f16(false, a, false, Bhh[0][kt],
                                                             (short)0, gr0, false, false);
                gr1 = __builtin_amdgcn_wmma_f32_16x16x32_f16(false, a, false, Bhh[1][kt],
                                                             (short)0, gr1, false, false);
            }
            // column-major writeback: 8 consecutive rows per lane -> ds_store_b128
            #pragma unroll
            for (int j = 0; j < 2; ++j) {
                const int jtx = jt0 + j;
                const v8f gi = j ? gi1 : gi0;
                const v8f gr = j ? gr1 : gr0;
                if (jtx < 16) {                  // r and z blocks: only gi+gr needed
                    float* dst = sS + (jtx * 16 + lo) * 16 + hi * 8;
                    v4f a0 = { gi[0] + gr[0], gi[1] + gr[1], gi[2] + gr[2], gi[3] + gr[3] };
                    v4f a1 = { gi[4] + gr[4], gi[5] + gr[5], gi[6] + gr[6], gi[7] + gr[7] };
                    *(v4f*)(dst)     = a0;
                    *(v4f*)(dst + 4) = a1;
                } else {                         // n block: keep gi, gr separate
                    float* di = sGiN + ((jtx - 16) * 16 + lo) * 16 + hi * 8;
                    float* dr = sGrN + ((jtx - 16) * 16 + lo) * 16 + hi * 8;
                    v4f i0 = { gi[0], gi[1], gi[2], gi[3] };
                    v4f i1 = { gi[4], gi[5], gi[6], gi[7] };
                    v4f r0 = { gr[0], gr[1], gr[2], gr[3] };
                    v4f r1 = { gr[4], gr[5], gr[6], gr[7] };
                    *(v4f*)(di)     = i0;
                    *(v4f*)(di + 4) = i1;
                    *(v4f*)(dr)     = r0;
                    *(v4f*)(dr + 4) = r1;
                }
            }
        }
        __syncthreads();   // sS / sGiN / sGrN ready

        // ---- gates & hidden update (column-major gate tiles: [j][m]) ----
        for (int i = tid; i < 16 * HH; i += NTHREADS) {
            int m = i >> 7, hh = i & 127;
            float rv = sigmoidf_fast(sS[hh * 16 + m] + sBih[hh] + sBhh[hh]);
            float zv = sigmoidf_fast(sS[(128 + hh) * 16 + m] + sBih[128 + hh] + sBhh[128 + hh]);
            float nv = tanhf_fast(sGiN[hh * 16 + m] + sBih[256 + hh] +
                                  rv * (sGrN[hh * 16 + m] + sBhh[256 + hh]));
            sH[i] = (1.f - zv) * nv + zv * sH[i];
        }

        // ---- output projection for prediction window: y = rep @ Wp^T + bp ----
        // fragP is loaded inside the branch (24 steps only) so it does not
        // occupy loop-invariant registers; L2/WGP$ resident after first use.
        if (t >= LH && wv < 4) {                 // wave-uniform branch: EXEC stays full
            v8f acc = {};
            #pragma unroll
            for (int kt = 0; kt < 4; ++kt) {
                v16h a  = *(const v16h*)(sA + (kt + 2) * 512 + lane * 16);  // decayed h
                v16h bf = *(const v16h*)(fragP + ((size_t)kt * 4 + wv) * 512 + lane * 16);
                acc = __builtin_amdgcn_wmma_f32_16x16x32_f16(false, a, false, bf,
                                                             (short)0, acc, false, false);
            }
            const int n = wv * 16 + lo;
            #pragma unroll
            for (int r = 0; r < 8; ++r) {
                int m = r + hi * 8;
                out[((size_t)(b0 + m) * LP + (t - LH)) * NF + n] = acc[r] + sBp[n];
            }
        }
        __syncthreads();   // protect sA/sH before next step rewrites them
    }
}

// ---------------------------------------------------------------------------
// Host-side launcher
// ---------------------------------------------------------------------------
extern "C" void kernel_launch(void* const* d_in, const int* in_sizes, int n_in,
                              void* d_out, int out_size, void* d_ws, size_t ws_size,
                              hipStream_t stream) {
    const float* tp_pred = (const float*)d_in[0];   // (B,24)
    const float* X       = (const float*)d_in[1];   // (B,96,64)
    const float* tp_true = (const float*)d_in[2];   // (B,96)
    const float* mask    = (const float*)d_in[3];   // (B,96,64)
    const float* Wh_dec  = (const float*)d_in[4];   // (128,64)
    const float* bh_dec  = (const float*)d_in[5];   // (128)
    const float* Wx_dec  = (const float*)d_in[6];   // (64,64)
    const float* bx_dec  = (const float*)d_in[7];   // (64)
    const float* W_ih    = (const float*)d_in[8];   // (384,256)
    const float* W_hh    = (const float*)d_in[9];   // (384,128)
    const float* b_ih    = (const float*)d_in[10];  // (384)
    const float* b_hh    = (const float*)d_in[11];  // (384)
    const float* Wp      = (const float*)d_in[12];  // (64,128)
    const float* bp      = (const float*)d_in[13];  // (64)
    float* out = (float*)d_out;                     // (1,B,24,64)

    // scratch layout (all offsets keep 32B alignment for v16h loads)
    float* ws_f      = (float*)d_ws;
    float* locf      = ws_f;                                  // B*120*64
    float* dtb       = locf + (size_t)BSZ * LALL * NF;        // B*120
    float* mu        = dtb + (size_t)BSZ * LALL;              // B*64
    float* rowsumWh  = mu + (size_t)BSZ * NF;                 // 128
    _Float16* fragIH = (_Float16*)(rowsumWh + HH);            // 8*24*512 halves
    _Float16* fragHH = fragIH + NIH;                          // 4*24*512
    _Float16* fragP  = fragHH + NHH;                          // 4*4*512

    grud_prep_kernel<<<BSZ, 128, 0, stream>>>(tp_pred, X, tp_true, mask,
                                              locf, dtb, mu);

    const int packTotal = NIH + NHH + NP + HH;
    grud_pack_kernel<<<(packTotal + 255) / 256, 256, 0, stream>>>(
        W_ih, W_hh, Wp, Wh_dec, fragIH, fragHH, fragP, rowsumWh);

    grud_main_kernel<<<BSZ / 16, NTHREADS, 0, stream>>>(
        X, mask, bh_dec, Wx_dec, bx_dec, b_ih, b_hh, bp,
        locf, dtb, mu, rowsumWh, fragIH, fragHH, fragP, out);
}
